// HierarchicalMemoryRouter_90726889160993
// MI455X (gfx1250) — matrix operand order repaired and checked by reading, not compiled, not run on hardware
//
#include <hip/hip_runtime.h>
#include <hip/hip_bf16.h>
#include <math.h>

typedef __attribute__((ext_vector_type(16))) _Float16 v16h;
typedef __attribute__((ext_vector_type(8)))  _Float16 v8h;
typedef __attribute__((ext_vector_type(8)))  float    v8f;

#define D_DIM   1024
#define S_SLOTS 96
#define NSSM    64
#define NT      6          // 96 / 16 N-tiles
#define MT      2          // M-tiles (16 rows each) per wave
#define KPANEL  128        // K panel cached in LDS
#define LSTR    136        // padded LDS row stride (halves): 272B rows
#define DECAY   0.9f

// ---------------------------------------------------------------------------
// Kernel 1: logits = input @ router_w^T (+b), row softmax, accumulate per-slot
// sums over T using V_WMMA_F32_16X16X32_F16.
// Block = 256 threads = 8 waves; each wave owns a 32-row strip (2 M-tiles)
// x 96 slots (6 N-tiles) -> 12 WMMAs per K-chunk, 1 ds_load_b128 per WMMA.
// ---------------------------------------------------------------------------
__global__ __launch_bounds__(256)
void router_softmax_kernel(const float* __restrict__ A,      // [T, 1024]
                           const float* __restrict__ rw,     // [96, 1024]
                           const float* __restrict__ rbias,  // [96]
                           float* __restrict__ sums,         // [96] (pre-zeroed)
                           int T) {
    __shared__ __align__(16) _Float16 ldsB[S_SLOTS * LSTR];

    const int tid  = threadIdx.x;
    const int wave = tid >> 5;
    const int lane = tid & 31;
    const int half = lane >> 4;   // 0: lanes 0-15, 1: lanes 16-31
    const int n    = lane & 15;   // N within tile / M within strip

    // bias per lane for each N-tile (slot = nt*16 + n)
    float biasv[NT];
#pragma unroll
    for (int nt = 0; nt < NT; ++nt) biasv[nt] = rbias[nt * 16 + n];

    float slotsum[NT];
#pragma unroll
    for (int nt = 0; nt < NT; ++nt) slotsum[nt] = 0.0f;

    const int rounds = T / (8 * MT * 16);   // 256 rows per workgroup round
    for (int round = blockIdx.x; round < rounds; round += gridDim.x) {
        const int r0 = round * (8 * MT * 16) + wave * (MT * 16);

        v8f acc[MT][NT];
#pragma unroll
        for (int mt = 0; mt < MT; ++mt)
#pragma unroll
            for (int nt = 0; nt < NT; ++nt) acc[mt][nt] = (v8f)0.0f;

        for (int panel = 0; panel < D_DIM / KPANEL; ++panel) {
            __syncthreads();
            // cooperative f32 -> f16 conversion of router_w panel into LDS
            for (int i = tid; i < S_SLOTS * KPANEL; i += 256) {
                const int s = i >> 7;           // /128
                const int c = i & (KPANEL - 1);
                ldsB[s * LSTR + c] = (_Float16)rw[(size_t)s * D_DIM + panel * KPANEL + c];
            }
            __syncthreads();

            // hint the next A panel into cache while we compute this one
            if (panel + 1 < D_DIM / KPANEL)
                __builtin_prefetch(A + (size_t)(r0 + n) * D_DIM + (panel + 1) * KPANEL + half * 8, 0, 0);

#pragma unroll
            for (int kc = 0; kc < KPANEL / 32; ++kc) {
                const int kg = panel * KPANEL + kc * 32;

                // --- A fragments (16x32 f16 each): lane half=0 -> K 0-15,
                //     half=1 -> K 16-31; M = lane&15 ---
                v16h a[MT];
#pragma unroll
                for (int mt = 0; mt < MT; ++mt) {
                    const float4* ap =
                        (const float4*)(A + (size_t)(r0 + mt * 16 + n) * D_DIM + kg + half * 16);
                    const float4 f0 = ap[0];   // K +0..3
                    const float4 f1 = ap[1];   // K +4..7
                    const float4 f2 = ap[2];   // K +8..11
                    const float4 f3 = ap[3];   // K +12..15
                    a[mt][0]  = (_Float16)f0.x; a[mt][1]  = (_Float16)f0.y;
                    a[mt][2]  = (_Float16)f0.z; a[mt][3]  = (_Float16)f0.w;
                    a[mt][4]  = (_Float16)f1.x; a[mt][5]  = (_Float16)f1.y;
                    a[mt][6]  = (_Float16)f1.z; a[mt][7]  = (_Float16)f1.w;
                    a[mt][8]  = (_Float16)f2.x; a[mt][9]  = (_Float16)f2.y;
                    a[mt][10] = (_Float16)f2.z; a[mt][11] = (_Float16)f2.w;
                    a[mt][12] = (_Float16)f3.x; a[mt][13] = (_Float16)f3.y;
                    a[mt][14] = (_Float16)f3.z; a[mt][15] = (_Float16)f3.w;
                }

#pragma unroll
                for (int nt = 0; nt < NT; ++nt) {
                    // --- B fragment (32x16 f16): N = lane&15,
                    //     lanes 0-15 hold K 0-15, lanes 16-31 hold K 16-31 ---
                    const v8h* bp =
                        (const v8h*)&ldsB[(nt * 16 + n) * LSTR + kc * 32 + half * 16];
                    const v8h b0 = bp[0];
                    const v8h b1 = bp[1];
                    v16h b;
#pragma unroll
                    for (int i = 0; i < 8; ++i) { b[i] = b0[i]; b[8 + i] = b1[i]; }

#pragma unroll
                    for (int mt = 0; mt < MT; ++mt) {
                        acc[mt][nt] = __builtin_amdgcn_wmma_f32_16x16x32_f16(
                            false, a[mt], false, b, (short)0, acc[mt][nt], false, false);
                    }
                }
            }
        }

        // --- per-row softmax over 96 slots; row = (mt, half, j),
        //     slot = nt*16 + lane%16 ---
#pragma unroll
        for (int mt = 0; mt < MT; ++mt) {
#pragma unroll
            for (int j = 0; j < 8; ++j) {
                float v[NT];
                float mx = -3.0e38f;
#pragma unroll
                for (int nt = 0; nt < NT; ++nt) {
                    v[nt] = acc[mt][nt][j] + biasv[nt];
                    mx = fmaxf(mx, v[nt]);
                }
#pragma unroll
                for (int m = 8; m >= 1; m >>= 1) mx = fmaxf(mx, __shfl_xor(mx, m, 32));
                float sm = 0.0f;
#pragma unroll
                for (int nt = 0; nt < NT; ++nt) { v[nt] = __expf(v[nt] - mx); sm += v[nt]; }
#pragma unroll
                for (int m = 8; m >= 1; m >>= 1) sm += __shfl_xor(sm, m, 32);
                const float inv = 1.0f / sm;
#pragma unroll
                for (int nt = 0; nt < NT; ++nt) slotsum[nt] += v[nt] * inv;
            }
        }
    }

    // fold the two halves together and emit one atomic per slot per wave
#pragma unroll
    for (int nt = 0; nt < NT; ++nt) {
        const float t = slotsum[nt] + __shfl_xor(slotsum[nt], 16, 32);
        if (lane < 16) atomicAdd(&sums[nt * 16 + lane], t);
    }
}

// ---------------------------------------------------------------------------
// Kernel 2: recent-mean, avg weights, weighted pooling, compress matvec,
// EMA state update with argmin eviction. Single block, 1024 threads.
// ---------------------------------------------------------------------------
__global__ __launch_bounds__(1024)
void finalize_kernel(const float* __restrict__ A,     // [T, 1024]
                     const float* __restrict__ ssm,   // [64, 1024]
                     const float* __restrict__ msm,   // [32, 1024]
                     const float* __restrict__ cw,    // [1024, 1024]
                     const float* __restrict__ cb,    // [1024]
                     const float* __restrict__ age,   // [64]
                     const float* __restrict__ sums,  // [96]
                     float* __restrict__ out,         // weighted|new_ssm|new_age
                     int T) {
    __shared__ float rm[D_DIM];
    __shared__ float tgt[D_DIM];
    __shared__ float aw[S_SLOTS];
    __shared__ int   idxmin;
    __shared__ int   evict;

    const int tid = threadIdx.x;

    // recent window mean over the last 8 rows
    float acc = 0.0f;
#pragma unroll
    for (int i = 0; i < 8; ++i) acc += A[(size_t)(T - 8 + i) * D_DIM + tid];
    rm[tid] = acc * 0.125f;

    if (tid < S_SLOTS) aw[tid] = sums[tid] / (float)T;
    __syncthreads();

    // target = compress_w @ recent_mean + compress_b
    float tg = cb[tid];
    const float* wrow = cw + (size_t)tid * D_DIM;
    for (int k = 0; k < D_DIM; ++k) tg = fmaf(wrow[k], rm[k], tg);
    tgt[tid] = tg;

    if (tid == 0) {
        int im = 0; float mv = aw[0];
        for (int s = 1; s < NSSM; ++s)
            if (aw[s] < mv) { mv = aw[s]; im = s; }   // first-min, matches argmin
        idxmin = im;
        float ma = age[0];
        for (int s = 1; s < NSSM; ++s) ma = fmaxf(ma, age[s]);
        evict = ((ma + 1.0f) > 50.0f) ? 1 : 0;
    }
    __syncthreads();

    // output 1: weighted = slots * avg_weights[:,None]
    for (int i = tid; i < S_SLOTS * D_DIM; i += 1024) {
        const int s = i >> 10;
        const int d = i & (D_DIM - 1);
        const float sv = (s < NSSM) ? ssm[i] : msm[(size_t)(s - NSSM) * D_DIM + d];
        out[i] = sv * aw[s];
    }

    // output 2: new_ssm EMA (+ eviction re-decay of argmin slot)
    float* out2 = out + S_SLOTS * D_DIM;
    const int im = idxmin;
    const int ev = evict;
    for (int i = tid; i < NSSM * D_DIM; i += 1024) {
        const int s = i >> 10;
        const int d = i & (D_DIM - 1);
        float v = DECAY * ssm[i] + (1.0f - DECAY) * tgt[d];
        if (ev && s == im) v = DECAY * v + (1.0f - DECAY) * tgt[d];
        out2[i] = v;
    }

    // output 3: new_age
    float* out3 = out2 + NSSM * D_DIM;
    if (tid < NSSM) {
        float a = age[tid] + 1.0f;
        if (ev && tid == im) a = 0.0f;
        out3[tid] = a;
    }
}

extern "C" void kernel_launch(void* const* d_in, const int* in_sizes, int n_in,
                              void* d_out, int out_size, void* d_ws, size_t ws_size,
                              hipStream_t stream) {
    const float* input = (const float*)d_in[0];
    const float* ssm   = (const float*)d_in[1];
    const float* msm   = (const float*)d_in[2];
    const float* rw    = (const float*)d_in[3];
    const float* rb    = (const float*)d_in[4];
    const float* cw    = (const float*)d_in[5];
    const float* cb    = (const float*)d_in[6];
    const float* age   = (const float*)d_in[7];
    float* out  = (float*)d_out;
    float* sums = (float*)d_ws;

    const int T = in_sizes[0] / D_DIM;

    hipMemsetAsync(sums, 0, S_SLOTS * sizeof(float), stream);

    const int rounds = T / (8 * MT * 16);   // 256 rows per workgroup round
    const int grid   = rounds < 1024 ? rounds : 1024;
    router_softmax_kernel<<<grid, 256, 0, stream>>>(input, rw, rb, sums, T);
    finalize_kernel<<<1, 1024, 0, stream>>>(input, ssm, msm, cw, cb, age, sums, out, T);
}
